// GraphEncoder_88064009437412
// MI455X (gfx1250) — matrix operand order repaired and checked by reading, not compiled, not run on hardware
//
#include <hip/hip_runtime.h>

typedef __attribute__((ext_vector_type(16))) _Float16 v16h;
typedef __attribute__((ext_vector_type(8)))  _Float16 v8h;
typedef __attribute__((ext_vector_type(4)))  _Float16 v4h;
typedef __attribute__((ext_vector_type(8)))  float    v8f;

#define NN 100000
#define EE 1600000
#define DD 100
#define KP 128             // padded K (row stride of packed f16 activations)
#define NP 112             // padded N for weight tiles (7*16)
#define MTILES (NN / 16)   // 6250, exact
#define NTILES 7           // ceil(100/16)

// ---------------- utility kernels ----------------

__global__ void zero_f32(float* __restrict__ p, int n) {
  int i = blockIdx.x * blockDim.x + threadIdx.x;
  if (i < n) p[i] = 0.0f;
}

__global__ void degree_k(const int* __restrict__ dst, float* __restrict__ deg, int E) {
  int e = blockIdx.x * blockDim.x + threadIdx.x;
  if (e < E) atomicAdd(&deg[dst[e]], 1.0f);
}

// f32 gather + f32 atomic scatter-add (layer-1 aggregation of x)
__global__ void scatter_add_f32(const float* __restrict__ x, const int* __restrict__ src,
                                const int* __restrict__ dst, float* __restrict__ agg, int E) {
  long long t = (long long)blockIdx.x * blockDim.x + threadIdx.x;
  if (t >= (long long)E * 25) return;
  int e = (int)(t / 25);
  int c = (int)(t % 25);
  int s = src[e], d = dst[e];
  const float4 v = reinterpret_cast<const float4*>(x + (long long)s * DD)[c];
  float* o = agg + (long long)d * DD + c * 4;
  atomicAdd(o + 0, v.x);
  atomicAdd(o + 1, v.y);
  atomicAdd(o + 2, v.z);
  atomicAdd(o + 3, v.w);
}

// f16 gather (packed [N][KP] rows) + f32 atomic scatter-add — half the gather bytes
__global__ void scatter_add_f16(const _Float16* __restrict__ xh, const int* __restrict__ src,
                                const int* __restrict__ dst, float* __restrict__ agg, int E) {
  long long t = (long long)blockIdx.x * blockDim.x + threadIdx.x;
  if (t >= (long long)E * 25) return;
  int e = (int)(t / 25);
  int c = (int)(t % 25);
  int s = src[e], d = dst[e];
  const v4h v = *reinterpret_cast<const v4h*>(xh + (long long)s * KP + c * 4);
  float* o = agg + (long long)d * DD + c * 4;
  atomicAdd(o + 0, (float)v[0]);
  atomicAdd(o + 1, (float)v[1]);
  atomicAdd(o + 2, (float)v[2]);
  atomicAdd(o + 3, (float)v[3]);
}

__global__ void normalize_k(float* __restrict__ agg, const float* __restrict__ deg, int n) {
  int i = blockIdx.x * blockDim.x + threadIdx.x;
  if (i < n) agg[i] = agg[i] / fmaxf(deg[i / DD], 1.0f);
}

// pack f32 [nrows][DD] -> f16 [nrows][KP], zero-padded in k
__global__ void pack_a_k(const float* __restrict__ in, _Float16* __restrict__ out, int nrows) {
  int idx = blockIdx.x * blockDim.x + threadIdx.x;
  if (idx >= nrows * KP) return;
  int i = idx >> 7, k = idx & (KP - 1);
  out[idx] = (k < DD) ? (_Float16)in[i * DD + k] : (_Float16)0.0f;
}

// pack f32 W[k][n] ([DD][DD]) -> f16 Wt[n][k] ([NP][KP]), zero-padded
__global__ void pack_w_k(const float* __restrict__ W, _Float16* __restrict__ Wt) {
  int idx = blockIdx.x * blockDim.x + threadIdx.x;
  if (idx >= NP * KP) return;
  int n = idx >> 7, k = idx & (KP - 1);
  Wt[idx] = (n < DD && k < DD) ? (_Float16)W[k * DD + n] : (_Float16)0.0f;
}

// ---------------- WMMA fragment loads (all guard-free, aligned) ----------------

// A 16x32 fragment: halves 0..15 <-> k = base+0..15 (contiguous)
__device__ __forceinline__ v16h load_a_frag(const _Float16* __restrict__ p) {
  v8h lo = *reinterpret_cast<const v8h*>(p);       // 16B aligned
  v8h hi = *reinterpret_cast<const v8h*>(p + 8);   // 16B aligned
  v16h a;
#pragma unroll
  for (int i = 0; i < 8; ++i) { a[i] = lo[i]; a[i + 8] = hi[i]; }
  return a;
}

// B 32x16 fragment: 16 contiguous halves of Wt row n starting at k = kb + kg16
__device__ __forceinline__ v16h load_b_frag(const _Float16* __restrict__ p) {
  return *reinterpret_cast<const v16h*>(p);        // 32B aligned
}

// ---------------- fused SAGE GEMM: out_s = Aagg@Wa_s + Aroot@Wr_s + b_s ----------------

template <int NSETS, bool RELU, bool OUT16>
__global__ void sage_gemm_k(const _Float16* __restrict__ Aagg, const _Float16* __restrict__ Aroot,
                            const _Float16* __restrict__ Wa0, const _Float16* __restrict__ Wr0,
                            const float* __restrict__ b0, void* __restrict__ out0v,
                            const _Float16* __restrict__ Wa1, const _Float16* __restrict__ Wr1,
                            const float* __restrict__ b1, float* __restrict__ out1) {
  const int wave = (blockIdx.x * blockDim.x + threadIdx.x) >> 5;
  const int lane = threadIdx.x & 31;
  if (wave >= MTILES * NTILES) return;
  const int mt = wave / NTILES;
  const int nt = wave % NTILES;

  const int khi8 = (lane & 16);       // 0 or 16 (A k-offset for upper half-wave)
  const int kg16 = (lane & 16);       // 0 or 16 (B k-group)
  const int rowA = mt * 16 + (lane & 15);
  const int n    = nt * 16 + (lane & 15);  // < 112 always

  const _Float16* pAa = Aagg  + (size_t)rowA * KP + khi8;
  const _Float16* pAr = Aroot + (size_t)rowA * KP + khi8;
  const _Float16* pWa0 = Wa0 + (size_t)n * KP + kg16;
  const _Float16* pWr0 = Wr0 + (size_t)n * KP + kg16;
  const _Float16* pWa1 = (NSETS == 2) ? Wa1 + (size_t)n * KP + kg16 : nullptr;
  const _Float16* pWr1 = (NSETS == 2) ? Wr1 + (size_t)n * KP + kg16 : nullptr;

  v8f c0 = {};
  v8f c1 = {};
#pragma unroll
  for (int kb = 0; kb < KP; kb += 32) {
    v16h aA = load_a_frag(pAa + kb);
    v16h aR = load_a_frag(pAr + kb);
    c0 = __builtin_amdgcn_wmma_f32_16x16x32_f16(false, aA, false,
            load_b_frag(pWa0 + kb), (short)0, c0, false, false);
    c0 = __builtin_amdgcn_wmma_f32_16x16x32_f16(false, aR, false,
            load_b_frag(pWr0 + kb), (short)0, c0, false, false);
    if (NSETS == 2) {
      c1 = __builtin_amdgcn_wmma_f32_16x16x32_f16(false, aA, false,
              load_b_frag(pWa1 + kb), (short)0, c1, false, false);
      c1 = __builtin_amdgcn_wmma_f32_16x16x32_f16(false, aR, false,
              load_b_frag(pWr1 + kb), (short)0, c1, false, false);
    }
  }

  if (n < DD) {  // single store-side guard; padded cols of f16 output stay zero
    const float bias0 = b0[n];
    const float bias1 = (NSETS == 2) ? b1[n] : 0.0f;
    const int mbase = mt * 16 + ((lane & 16) >> 1);  // C layout: VGPR r <-> M = r (+8 upper)
#pragma unroll
    for (int r = 0; r < 8; ++r) {
      const long long m = mbase + r;
      float v0 = c0[r] + bias0;
      if (RELU) v0 = fmaxf(v0, 0.0f);
      if (OUT16) {
        ((_Float16*)out0v)[m * KP + n] = (_Float16)v0;
      } else {
        ((float*)out0v)[m * DD + n] = v0;
      }
      if (NSETS == 2) out1[m * DD + n] = c1[r] + bias1;
    }
  }
}

// ---------------- launcher ----------------

extern "C" void kernel_launch(void* const* d_in, const int* in_sizes, int n_in,
                              void* d_out, int out_size, void* d_ws, size_t ws_size,
                              hipStream_t stream) {
  const float* x   = (const float*)d_in[0];
  const int*   ei  = (const int*)d_in[1];
  // d_in[2] edge_weight: accepted but unused (matches reference)
  const float* Wsa = (const float*)d_in[3];
  const float* Wsr = (const float*)d_in[4];
  const float* bs  = (const float*)d_in[5];
  const float* Wma = (const float*)d_in[6];
  const float* Wmr = (const float*)d_in[7];
  const float* bm  = (const float*)d_in[8];
  const float* Wva = (const float*)d_in[9];
  const float* Wvr = (const float*)d_in[10];
  const float* bv  = (const float*)d_in[11];

  const int* src = ei;
  const int* dst = ei + EE;

  // workspace layout (bytes, all 32B-aligned):
  // [deg N*f32][agg N*DD*f32][feath N*KP*f16][xh N*KP*f16][aggh N*KP*f16][Wp 6*NP*KP*f16]
  char* ws = (char*)d_ws;
  float*    deg   = (float*)ws;                                     // 0.4 MB
  float*    agg   = (float*)(ws + (size_t)NN * 4);                  // 40 MB (aggX then aggF)
  _Float16* feath = (_Float16*)(ws + (size_t)NN * 4 + (size_t)NN * DD * 4);
  _Float16* xh    = feath + (size_t)NN * KP;
  _Float16* aggh  = xh    + (size_t)NN * KP;
  _Float16* Wp    = aggh  + (size_t)NN * KP;                        // 6 x NP*KP
  _Float16* Wt_sa = Wp + 0 * (NP * KP);
  _Float16* Wt_sr = Wp + 1 * (NP * KP);
  _Float16* Wt_ma = Wp + 2 * (NP * KP);
  _Float16* Wt_mr = Wp + 3 * (NP * KP);
  _Float16* Wt_va = Wp + 4 * (NP * KP);
  _Float16* Wt_vr = Wp + 5 * (NP * KP);

  float* mu  = (float*)d_out;
  float* var = (float*)d_out + (size_t)NN * DD;

  const int B = 256;

  // zero deg + agg + feath in one contiguous pass (feath pad cols must be 0)
  const int zn = NN + NN * DD + NN * (KP / 2);  // f16 region counted as f32 pairs
  zero_f32<<<(zn + B - 1) / B, B, 0, stream>>>((float*)ws, zn);

  // pack the 6 weight matrices into transposed, padded f16 tiles
  const int wb = (NP * KP + B - 1) / B;
  pack_w_k<<<wb, B, 0, stream>>>(Wsa, Wt_sa);
  pack_w_k<<<wb, B, 0, stream>>>(Wsr, Wt_sr);
  pack_w_k<<<wb, B, 0, stream>>>(Wma, Wt_ma);
  pack_w_k<<<wb, B, 0, stream>>>(Wmr, Wt_mr);
  pack_w_k<<<wb, B, 0, stream>>>(Wva, Wt_va);
  pack_w_k<<<wb, B, 0, stream>>>(Wvr, Wt_vr);

  degree_k<<<(EE + B - 1) / B, B, 0, stream>>>(dst, deg, EE);

  const long long sc = (long long)EE * 25;
  const unsigned scBlocks = (unsigned)((sc + B - 1) / B);
  const int nd  = NN * DD;
  const int ndB = (nd + B - 1) / B;
  const int pk  = (NN * KP + B - 1) / B;

  // layer-1 aggregation (f32 gather of x)
  scatter_add_f32<<<scBlocks, B, 0, stream>>>(x, src, dst, agg, EE);
  normalize_k<<<ndB, B, 0, stream>>>(agg, deg, nd);

  // pack A operands for layer 1
  pack_a_k<<<pk, B, 0, stream>>>(agg, aggh, NN);
  pack_a_k<<<pk, B, 0, stream>>>(x, xh, NN);

  const int waves  = MTILES * NTILES;   // 43750 tiles
  const int blocks = (waves + 3) / 4;   // 4 waves / 128-thread block

  // layer 1: feath = relu(aggX@Wsa + x@Wsr + bs), written directly as packed f16
  sage_gemm_k<1, true, true><<<blocks, 128, 0, stream>>>(
      aggh, xh, Wt_sa, Wt_sr, bs, (void*)feath,
      nullptr, nullptr, nullptr, nullptr);

  // shared aggregation of feat for mu and var (f16 gather, f32 atomics)
  zero_f32<<<ndB, B, 0, stream>>>(agg, nd);
  scatter_add_f16<<<scBlocks, B, 0, stream>>>(feath, src, dst, agg, EE);
  normalize_k<<<ndB, B, 0, stream>>>(agg, deg, nd);
  pack_a_k<<<pk, B, 0, stream>>>(agg, aggh, NN);

  // layers 2+3 fused: mu = aggF@Wma + feat@Wmr + bm ; var = aggF@Wva + feat@Wvr + bv
  sage_gemm_k<2, false, false><<<blocks, 128, 0, stream>>>(
      aggh, feath, Wt_ma, Wt_mr, bm, (void*)mu,
      Wt_va, Wt_vr, bv, var);
}
